// GAT_81192061764228
// MI455X (gfx1250) — matrix operand order repaired and checked by reading, not compiled, not run on hardware
//
#include <hip/hip_runtime.h>

typedef __attribute__((ext_vector_type(16))) _Float16 v16h;
typedef __attribute__((ext_vector_type(8)))  _Float16 v8h;
typedef __attribute__((ext_vector_type(8)))  float    v8f;
typedef __attribute__((ext_vector_type(4)))  float    v4f;
typedef __attribute__((ext_vector_type(4)))  unsigned u32x4;
typedef __attribute__((ext_vector_type(8)))  int      i32x8;
typedef __attribute__((ext_vector_type(4)))  int      i32x4;

static constexpr int B_  = 1024;
static constexpr int N_  = 64;
static constexpr int F_  = 1024;
static constexpr int H1_ = 256;
static constexpr int H2_ = 128;
static constexpr int NC_ = 128;
#define ALPHA_  0.2f
#define BN_EPS_ 1e-5f

#ifndef __has_builtin
#define __has_builtin(x) 0
#endif
#if __has_builtin(__builtin_amdgcn_tensor_load_to_lds) && \
    __has_builtin(__builtin_amdgcn_s_wait_tensorcnt)
#  define TDM_AVAILABLE 1
#  if __has_include(<hip/amd_detail/amd_gfx1250_TDM.h>)
#    define TDM_6ARG 1
#  else
#    define TDM_6ARG 0
#  endif
#else
#  define TDM_AVAILABLE 0
#endif

// l2norm over singleton axis == t / max(|t|, 1e-12) == sign(t) except tiny values
__device__ __forceinline__ float l2n(float v) {
    float a = fabsf(v);
    return (a >= 1e-12f) ? copysignf(1.0f, v) : (v * 1e12f);
}

// ---------------- prep kernels ----------------
__global__ void k_cvt_f16(const float* __restrict__ src, _Float16* __restrict__ dst, int n) {
    int i = blockIdx.x * blockDim.x + threadIdx.x;
    if (i < n) dst[i] = (_Float16)src[i];
}

// va[f] = sum_h W1[h,f]*a11[h];  vb[f] = sum_h W1[h,f]*a12[h]
__global__ void k_prep_vab(const float* __restrict__ W1, const float* __restrict__ a11,
                           const float* __restrict__ a12, float* __restrict__ va,
                           float* __restrict__ vb) {
    int f = blockIdx.x * blockDim.x + threadIdx.x;
    float sa = 0.f, sb = 0.f;
    for (int h = 0; h < H1_; ++h) {
        float w = W1[(size_t)h * F_ + f];
        sa += w * a11[h];
        sb += w * a12[h];
    }
    va[f] = sa; vb[f] = sb;
}

// v2a[h1] = sum_h2 W2[h2,h1]*a21[h2]; v2b likewise with a22
__global__ void k_prep_v2(const float* __restrict__ W2, const float* __restrict__ a21,
                          const float* __restrict__ a22, float* __restrict__ v2a,
                          float* __restrict__ v2b) {
    int h1 = threadIdx.x; // 256 threads
    float sa = 0.f, sb = 0.f;
    for (int h2 = 0; h2 < H2_; ++h2) {
        float w = W2[(size_t)h2 * H1_ + h1];
        sa += w * a21[h2];
        sb += w * a22[h2];
    }
    v2a[h1] = sa; v2b[h1] = sb;
}

// consts[0] = b1.(a11+a12), consts[1] = b2.(a21+a22)
__global__ void k_prep_c(const float* __restrict__ b1, const float* __restrict__ a11,
                         const float* __restrict__ a12, const float* __restrict__ b2,
                         const float* __restrict__ a21, const float* __restrict__ a22,
                         float* __restrict__ consts) {
    __shared__ float r1[256], r2[256];
    int t = threadIdx.x;
    r1[t] = b1[t] * (a11[t] + a12[t]);
    r2[t] = (t < H2_) ? b2[t] * (a21[t] + a22[t]) : 0.f;
    __syncthreads();
    for (int s = 128; s > 0; s >>= 1) {
        if (t < s) { r1[t] += r1[t + s]; r2[t] += r2[t + s]; }
        __syncthreads();
    }
    if (t == 0) { consts[0] = r1[0]; consts[1] = r2[0]; }
}

// ---------------- layer-1 attention: scores -> softmax -> aggregate ----------------
__global__ void k_attn1(const float* __restrict__ x, const float* __restrict__ nb,
                        const float* __restrict__ va_g, const float* __restrict__ vb_g,
                        const float* __restrict__ consts, _Float16* __restrict__ agg1) {
    __shared__ float vaS[F_], vbS[F_];
    __shared__ float sS[N_];
    __shared__ float sxS, invS;
    int b = blockIdx.x, t = threadIdx.x;
    for (int i = t; i < F_; i += 256) { vaS[i] = va_g[i]; vbS[i] = vb_g[i]; }
    if (t == 0) sxS = 0.f;
    __syncthreads();
    int wave = t >> 5, lane = t & 31;
    // neighbor scores: each wave handles 8 neighbors
    for (int j = 0; j < 8; ++j) {
        int n = wave * 8 + j;
        const float* row = nb + ((size_t)b * N_ + n) * F_;
        float p = 0.f;
        for (int k = 0; k < F_ / 32; ++k) {
            int f = lane + 32 * k;
            p += l2n(row[f]) * vbS[f];
        }
        for (int m = 16; m; m >>= 1) p += __shfl_xor(p, m, 32);
        if (lane == 0) sS[n] = p;
    }
    // x score: wave w covers f in [w*128, w*128+128)
    {
        const float* xr = x + (size_t)b * F_;
        float p = 0.f;
        for (int k = 0; k < 4; ++k) {
            int f = wave * 128 + lane + 32 * k;
            p += l2n(xr[f]) * vaS[f];
        }
        for (int m = 16; m; m >>= 1) p += __shfl_xor(p, m, 32);
        if (lane == 0) atomicAdd(&sxS, p);
    }
    __syncthreads();
    if (t == 0) {
        float sx = sxS + consts[0];
        float mx = -1e30f;
        for (int n = 0; n < N_; ++n) {
            float s = sx + sS[n];
            float l = s > 0.f ? s : ALPHA_ * s;   // LeakyReLU
            sS[n] = l; mx = fmaxf(mx, l);
        }
        float sum = 0.f;
        for (int n = 0; n < N_; ++n) { float e = expf(sS[n] - mx); sS[n] = e; sum += e; }
        invS = 1.f / sum;
    }
    __syncthreads();
    float inv = invS;
    for (int c = 0; c < 4; ++c) {
        int f = t + 256 * c;
        const float* base = nb + (size_t)b * N_ * F_ + f;
        float acc = 0.f;
        for (int n = 0; n < N_; ++n) acc += sS[n] * l2n(base[(size_t)n * F_]);
        agg1[(size_t)b * F_ + f] = (_Float16)(acc * inv);
    }
}

// ---------------- WMMA GEMM: D[M,256] = A[M,1024] @ W1h^T + bias ----------------
// One block = 64 rows. 8 waves: wave w -> mtile = w>>1, ntiles [(w&1)*8, +8).
// B panels (32K x 256N, f16) staged into LDS by the Tensor Data Mover
// (double-buffered, TENSORcnt-tracked), padded to an 80B row stride.
template <bool NORMA, bool STATS, bool OUT16>
__global__ void k_gemm(const void* __restrict__ Asrc, const _Float16* __restrict__ Bw,
                       const float* __restrict__ bias, void* __restrict__ Dout,
                       float* __restrict__ stats) {
    __shared__ float biasS[H1_];
    __shared__ float redS[256], redS2[256];
#if TDM_AVAILABLE
    constexpr int BROW = 40;                   // 32 data halfs + 8 pad halfs = 80B
    __shared__ _Float16 ldsB[2][H1_ * BROW];   // 2 x 20KB B panels
#endif
    int t = threadIdx.x;
    if (t < H1_) biasS[t] = bias[t];
    __syncthreads();

    int wave = t >> 5, lane = t & 31;
    int mtile = wave >> 1;
    int ntbase = (wave & 1) * 8;
    int mrow0 = blockIdx.x * 64 + mtile * 16;
    int m = lane & 15, half = lane >> 4;

    v8f acc[8];
    v8f zero = {0.f, 0.f, 0.f, 0.f, 0.f, 0.f, 0.f, 0.f};
#pragma unroll
    for (int i = 0; i < 8; ++i) acc[i] = zero;

#if TDM_AVAILABLE
    unsigned ldsbase = (unsigned)(unsigned long long)(uintptr_t)(&ldsB[0][0]);
    auto issue_panel = [&](int ks, int buf) {
        // TDM D# (ISA 8.3/8.4): 2-D tile, data_size=2B, 4-dword pad per 16 dwords
        unsigned long long ga =
            (unsigned long long)(uintptr_t)Bw + (unsigned long long)ks * 64ull;
        u32x4 g0;
        g0[0] = 1u;                                        // count=1, no gather
        g0[1] = ldsbase + (unsigned)buf * (unsigned)(H1_ * BROW * 2); // lds_addr
        g0[2] = (unsigned)ga;                              // global_addr[31:0]
        g0[3] = (unsigned)((ga >> 32) & 0x1FFFFFFull) | (2u << 30); // addr[56:32],type=2
        i32x8 g1;
        g1[0] = (1 << 16) | (1 << 20) | (3 << 22) | (3 << 25); // 2B, pad 4dw/16dw
        g1[1] = (int)((F_ & 0xFFFF) << 16);                // tensor_dim0 = 1024 (lo16)
        g1[2] = (int)((F_ >> 16) | (H1_ << 16));           // dim0 hi, tensor_dim1 = 256
        g1[3] = (int)((H1_ >> 16) | (32 << 16));           // dim1 hi, tile_dim0 = 32
        g1[4] = H1_;                                       // tile_dim1 = 256, tile_dim2=0
        g1[5] = F_;                                        // tensor_dim0_stride = 1024
        g1[6] = 0;
        g1[7] = 0;
        i32x4 z4 = {0, 0, 0, 0};
#if TDM_6ARG
        i32x8 z8 = {0, 0, 0, 0, 0, 0, 0, 0};
        __builtin_amdgcn_tensor_load_to_lds(g0, g1, z4, z4, z8, 0);
#else
        __builtin_amdgcn_tensor_load_to_lds(g0, g1, z4, z4, 0);
#endif
    };
    if (wave == 0) issue_panel(0, 0);
#endif

    for (int ks = 0; ks < F_ / 32; ++ks) {
#if TDM_AVAILABLE
        if (wave == 0) {
            if (ks + 1 < F_ / 32) {
                issue_panel(ks + 1, (ks + 1) & 1);
                __builtin_amdgcn_s_wait_tensorcnt(1);  // panel ks complete (in-order)
            } else {
                __builtin_amdgcn_s_wait_tensorcnt(0);
            }
        }
        __syncthreads();  // publish panel ks to all waves
#endif
        // A fragment: 16-bit A 16x32 layout (ISA 7.12.2):
        // lanes 0-15: K = [2j,2j+1] & [16..23]; lanes 16-31: +8
        int kbaseA = ks * 32 + 8 * half;
        v16h afrag;
        if constexpr (NORMA) {
            const float* arow = ((const float*)Asrc) + (size_t)(mrow0 + m) * F_;
            const v4f* c0 = (const v4f*)(arow + kbaseA);
            const v4f* c1 = (const v4f*)(arow + kbaseA + 16);
            v4f x0 = c0[0], x1 = c0[1], y0 = c1[0], y1 = c1[1];
#pragma unroll
            for (int i = 0; i < 4; ++i) {
                afrag[i]      = (_Float16)l2n(x0[i]);
                afrag[4 + i]  = (_Float16)l2n(x1[i]);
                afrag[8 + i]  = (_Float16)l2n(y0[i]);
                afrag[12 + i] = (_Float16)l2n(y1[i]);
            }
        } else {
            const _Float16* arow = ((const _Float16*)Asrc) + (size_t)(mrow0 + m) * F_;
            v8h h0 = *(const v8h*)(arow + kbaseA);
            v8h h1 = *(const v8h*)(arow + kbaseA + 16);
#pragma unroll
            for (int i = 0; i < 8; ++i) { afrag[i] = h0[i]; afrag[8 + i] = h1[i]; }
        }
        // B fragment: lanes 0-15 hold K=0..7,16..23; lanes 16-31 hold K=8..15,24..31 (column = lane&15)
#pragma unroll
        for (int nt = 0; nt < 8; ++nt) {
            int ncol = (ntbase + nt) * 16 + m;
            v8h b0, b1;
#if TDM_AVAILABLE
            const _Float16* brow = &ldsB[ks & 1][ncol * BROW + 8 * half];
            b0 = *(const v8h*)brow;
            b1 = *(const v8h*)(brow + 16);
#else
            const _Float16* brow = Bw + (size_t)ncol * F_ + (ks * 32 + 8 * half);
            b0 = *(const v8h*)brow;
            b1 = *(const v8h*)(brow + 16);
#endif
            v16h bfrag;
#pragma unroll
            for (int i = 0; i < 8; ++i) { bfrag[i] = b0[i]; bfrag[8 + i] = b1[i]; }
            acc[nt] = __builtin_amdgcn_wmma_f32_16x16x32_f16(
                false, afrag, false, bfrag, (short)0, acc[nt], false, false);
        }
#if TDM_AVAILABLE
        __syncthreads();  // all waves done with panel ks before its buffer is rewritten
#endif
    }

    // epilogue: +bias, optional per-block (per-b) BN stats, store
    float ssum = 0.f, ssq = 0.f;
#pragma unroll
    for (int nt = 0; nt < 8; ++nt) {
        int ncol = (ntbase + nt) * 16 + m;
        float bv = biasS[ncol];
#pragma unroll
        for (int r = 0; r < 8; ++r) {
            int row = mrow0 + r + 8 * half;  // D layout: VGPR r -> M=r (lanes<16) / r+8
            float v = acc[nt][r] + bv;
            if (STATS) { ssum += v; ssq += v * v; }
            if (OUT16) ((_Float16*)Dout)[(size_t)row * H1_ + ncol] = (_Float16)v;
            else       ((float*)Dout)[(size_t)row * H1_ + ncol] = v;
        }
    }
    if (STATS) {
        redS[t] = ssum; redS2[t] = ssq;
        __syncthreads();
        for (int s = 128; s > 0; s >>= 1) {
            if (t < s) { redS[t] += redS[t + s]; redS2[t] += redS2[t + s]; }
            __syncthreads();
        }
        if (t == 0) {
            float mean = redS[0] * (1.f / 16384.f);
            float var = redS2[0] * (1.f / 16384.f) - mean * mean;
            stats[blockIdx.x * 2] = mean;
            stats[blockIdx.x * 2 + 1] = rsqrtf(var + BN_EPS_);
        }
    }
}

// ---------------- global BN stats (single block) ----------------
__global__ void k_gstats(const float* __restrict__ src, int n, float* __restrict__ consts,
                         int outIdx) {
    __shared__ float r1[1024], r2[1024];
    int t = threadIdx.x;
    float s = 0.f, q = 0.f;
    for (int i = t; i < n; i += 1024) { float v = src[i]; s += v; q += v * v; }
    r1[t] = s; r2[t] = q;
    __syncthreads();
    for (int st = 512; st > 0; st >>= 1) {
        if (t < st) { r1[t] += r1[t + st]; r2[t] += r2[t + st]; }
        __syncthreads();
    }
    if (t == 0) {
        float mean = r1[0] / (float)n;
        float var = r2[0] / (float)n - mean * mean;
        consts[outIdx] = mean;
        consts[outIdx + 1] = rsqrtf(var + BN_EPS_);
    }
}

// ---------------- layer 2 fused: BN+ReLU, scores, softmax, aggregate, GEMM ----------------
__global__ void k_layer2(const float* __restrict__ out1, const _Float16* __restrict__ nbh,
                         const float* __restrict__ statsB, const float* __restrict__ consts,
                         const float* __restrict__ v2a, const float* __restrict__ v2b,
                         const float* __restrict__ W2, const float* __restrict__ b2,
                         const float* __restrict__ g1p, const float* __restrict__ be1p,
                         float* __restrict__ out2) {
    __shared__ _Float16 nb2S[N_ * H1_];  // 32KB
    __shared__ float x2S[H1_], aggS[H1_], sS[N_], red[256];
    __shared__ float sxS, invS;
    int b = blockIdx.x, t = threadIdx.x;
    float g1 = g1p[0], be1 = be1p[0];
    float m1 = consts[2], r1 = consts[3];
    float mb = statsB[b * 2], rb = statsB[b * 2 + 1];
    {
        float v = out1[(size_t)b * H1_ + t];
        float xv = fmaxf(g1 * (v - m1) * r1 + be1, 0.f);
        x2S[t] = xv;
        red[t] = xv * v2a[t];
    }
    for (int i = t; i < N_ * H1_; i += 256) {
        float v = (float)nbh[(size_t)b * (N_ * H1_) + i];
        nb2S[i] = (_Float16)fmaxf(g1 * (v - mb) * rb + be1, 0.f);
    }
    __syncthreads();
    for (int s = 128; s > 0; s >>= 1) {
        if (t < s) red[t] += red[t + s];
        __syncthreads();
    }
    if (t == 0) sxS = red[0];
    int wave = t >> 5, lane = t & 31;
    for (int j = 0; j < 8; ++j) {
        int n = wave * 8 + j;
        float p = 0.f;
        for (int k = 0; k < 8; ++k) {
            int h = lane + 32 * k;
            p += (float)nb2S[n * H1_ + h] * v2b[h];
        }
        for (int m = 16; m; m >>= 1) p += __shfl_xor(p, m, 32);
        if (lane == 0) sS[n] = p;
    }
    __syncthreads();
    if (t == 0) {
        float sx = sxS + consts[1];
        float mx = -1e30f;
        for (int n = 0; n < N_; ++n) {
            float s = sx + sS[n];
            float l = s > 0.f ? s : ALPHA_ * s;
            sS[n] = l; mx = fmaxf(mx, l);
        }
        float sum = 0.f;
        for (int n = 0; n < N_; ++n) { float e = expf(sS[n] - mx); sS[n] = e; sum += e; }
        invS = 1.f / sum;
    }
    __syncthreads();
    {
        float inv = invS, a = 0.f;
        for (int n = 0; n < N_; ++n) a += sS[n] * (float)nb2S[n * H1_ + t];
        aggS[t] = a * inv;
    }
    __syncthreads();
    if (t < H2_) {
        float acc = b2[t];
        const float* wr = W2 + (size_t)t * H1_;
        for (int h = 0; h < H1_; ++h) acc += aggS[h] * wr[h];
        out2[(size_t)b * H2_ + t] = acc;
    }
}

// ---------------- classifier ----------------
__global__ void k_logits(const float* __restrict__ out2, const float* __restrict__ Wl,
                         const float* __restrict__ bl, const float* __restrict__ consts,
                         const float* __restrict__ g2p, const float* __restrict__ be2p,
                         float* __restrict__ logits) {
    __shared__ float x3S[H2_];
    int b = blockIdx.x, t = threadIdx.x; // 128 threads
    float g2 = g2p[0], be2 = be2p[0], m2 = consts[4], r2 = consts[5];
    float v = out2[(size_t)b * H2_ + t];
    x3S[t] = fmaxf(g2 * (v - m2) * r2 + be2, 0.f);
    __syncthreads();
    float acc = bl[t];
    const float* wr = Wl + (size_t)t * H2_;
    for (int h = 0; h < H2_; ++h) acc += x3S[h] * wr[h];
    logits[(size_t)b * NC_ + t] = acc;
}

extern "C" void kernel_launch(void* const* d_in, const int* in_sizes, int n_in,
                              void* d_out, int out_size, void* d_ws, size_t ws_size,
                              hipStream_t stream) {
    const float* x   = (const float*)d_in[0];
    const float* nb  = (const float*)d_in[1];
    const float* W1  = (const float*)d_in[2];
    const float* b1  = (const float*)d_in[3];
    const float* a11 = (const float*)d_in[4];
    const float* a12 = (const float*)d_in[5];
    const float* g1  = (const float*)d_in[6];
    const float* be1 = (const float*)d_in[7];
    const float* W2  = (const float*)d_in[8];
    const float* b2  = (const float*)d_in[9];
    const float* a21 = (const float*)d_in[10];
    const float* a22 = (const float*)d_in[11];
    const float* g2  = (const float*)d_in[12];
    const float* be2 = (const float*)d_in[13];
    const float* Wl  = (const float*)d_in[14];
    const float* bl  = (const float*)d_in[15];
    float* logits = (float*)d_out;

    // workspace layout (256B aligned)
    char* ws = (char*)d_ws;
    size_t o = 0;
    auto alloc = [&](size_t bytes) { size_t r = o; o += (bytes + 255) & ~(size_t)255; return r; };
    _Float16* W1h   = (_Float16*)(ws + alloc((size_t)H1_ * F_ * 2));     // 512KB
    float*    va    = (float*)(ws + alloc((size_t)F_ * 4));
    float*    vb    = (float*)(ws + alloc((size_t)F_ * 4));
    float*    v2a   = (float*)(ws + alloc((size_t)H1_ * 4));
    float*    v2b   = (float*)(ws + alloc((size_t)H1_ * 4));
    float*    consts= (float*)(ws + alloc(64 * 4));
    _Float16* agg1  = (_Float16*)(ws + alloc((size_t)B_ * F_ * 2));      // 2MB
    _Float16* nbh   = (_Float16*)(ws + alloc((size_t)B_ * N_ * H1_ * 2));// 32MB
    float*    statsB= (float*)(ws + alloc((size_t)B_ * 2 * 4));
    float*    out1  = (float*)(ws + alloc((size_t)B_ * H1_ * 4));        // 1MB
    float*    out2  = (float*)(ws + alloc((size_t)B_ * H2_ * 4));        // 512KB
    (void)ws_size; (void)n_in; (void)in_sizes; (void)out_size;

    // prep
    k_cvt_f16<<<dim3((H1_ * F_) / 256), dim3(256), 0, stream>>>(W1, W1h, H1_ * F_);
    k_prep_vab<<<dim3(F_ / 256), dim3(256), 0, stream>>>(W1, a11, a12, va, vb);
    k_prep_v2<<<dim3(1), dim3(256), 0, stream>>>(W2, a21, a22, v2a, v2b);
    k_prep_c<<<dim3(1), dim3(256), 0, stream>>>(b1, a11, a12, b2, a21, a22, consts);

    // layer 1 attention + aggregate (x branch only influences attention scores)
    k_attn1<<<dim3(B_), dim3(256), 0, stream>>>(x, nb, va, vb, consts, agg1);

    // nbh = norm(nb) @ W1^T + b1  (f16 WMMA + TDM-staged B, per-b BN stats fused) -> f16
    k_gemm<true, true, true><<<dim3(B_), dim3(256), 0, stream>>>(nb, W1h, b1, nbh, statsB);
    // out1 = agg1 @ W1^T + b1 -> f32
    k_gemm<false, false, false><<<dim3(B_ / 16), dim3(256), 0, stream>>>(agg1, W1h, b1, out1, nullptr);

    // global BN stats over out1
    k_gstats<<<dim3(1), dim3(1024), 0, stream>>>(out1, B_ * H1_, consts, 2);

    // layer 2 fused
    k_layer2<<<dim3(B_), dim3(256), 0, stream>>>(out1, nbh, statsB, consts, v2a, v2b,
                                                 W2, b2, g1, be1, out2);

    // global BN stats over out2, then classifier
    k_gstats<<<dim3(1), dim3(1024), 0, stream>>>(out2, B_ * H2_, consts, 4);
    k_logits<<<dim3(B_), dim3(NC_), 0, stream>>>(out2, Wl, bl, consts, g2, be2, logits);
}
